// BangaloreLSTM_9311489097866
// MI455X (gfx1250) — matrix operand order repaired and checked by reading, not compiled, not run on hardware
//
#include <hip/hip_runtime.h>

// ---------------------------------------------------------------------------
// Fused 2-layer LSTM + FC head for MI455X (gfx1250, wave32, WMMA).
// One workgroup (256 thr = 8 waves) owns 16 batch rows for all 168 timesteps.
// Weights are staged fp32->fp16 in LDS once, then the per-wave WMMA B-matrix
// fragments are hoisted into registers (loop-invariant across timesteps).
// Accumulator chains start from WMMA's inline-zero C operand; the gate bias
// is folded in as a scalar v_add at store time (keeps VGPR count low).
// ---------------------------------------------------------------------------

typedef __attribute__((ext_vector_type(16))) _Float16 v16h;
typedef __attribute__((ext_vector_type(8)))  float    v8f;

#define B_TOT 4096
#define T_LEN 168
#define D_IN  16
#define H_DIM 96
#define G4    384   // 4*H

union FragU { v16h v; unsigned u[8]; };

// A-matrix 16x32 f16 fragment (ISA 7.12.2): lane holds row M=lane&15,
// lane-half kg selects K-half; VGPR v holds K pairs {0,2,4,6,16,18,20,22}+8*kg.
__device__ __forceinline__ v16h ld_a(const _Float16* p, int kg) {
    FragU f;
    const int ka[8] = {0, 2, 4, 6, 16, 18, 20, 22};
#pragma unroll
    for (int v = 0; v < 8; ++v)
        f.u[v] = *reinterpret_cast<const unsigned*>(p + ka[v] + 8 * kg);
    return f.v;
}

// B-matrix 32x16 f16 fragment: lane holds col N=lane&15, lane-half kg selects
// K-half; VGPR v holds K pairs {0,2,4,6,16,18,20,22}+8*kg. p points at row n, k0.
__device__ __forceinline__ v16h ld_b(const _Float16* p, int kg) {
    FragU f;
    const int ka[8] = {0, 2, 4, 6, 16, 18, 20, 22};
#pragma unroll
    for (int v = 0; v < 8; ++v)
        f.u[v] = *reinterpret_cast<const unsigned*>(p + ka[v] + 8 * kg);
    return f.v;
}

__device__ __forceinline__ v8f wmma16(v16h a, v16h b, v8f c) {
    return __builtin_amdgcn_wmma_f32_16x16x32_f16(
        /*neg_a=*/false, a, /*neg_b=*/false, b,
        /*c_mod=*/(short)0, c, /*reuse_a=*/false, /*reuse_b=*/false);
}

__device__ __forceinline__ float fast_sigmoid(float x) {
    return __builtin_amdgcn_rcpf(1.0f + __expf(-x));
}
__device__ __forceinline__ float fast_tanh(float x) {
    return 2.0f * __builtin_amdgcn_rcpf(1.0f + __expf(-2.0f * x)) - 1.0f;
}

__global__ void __launch_bounds__(256, 1)
BangaloreLSTM_kernel(const float* __restrict__ x,
                     const float* __restrict__ Wih0, const float* __restrict__ Whh0,
                     const float* __restrict__ bih0, const float* __restrict__ bhh0,
                     const float* __restrict__ Wih1, const float* __restrict__ Whh1,
                     const float* __restrict__ bih1, const float* __restrict__ bhh1,
                     const float* __restrict__ Wfc,  const float* __restrict__ bfc,
                     float* __restrict__ out)
{
    extern __shared__ char smem_raw[];
    _Float16* s_wih0 = (_Float16*)smem_raw;            // 384 x 32 (K padded, zero top)
    _Float16* s_whh0 = s_wih0 + G4 * 32;               // 384 x 96
    _Float16* s_wih1 = s_whh0 + G4 * H_DIM;            // 384 x 96
    _Float16* s_whh1 = s_wih1 + G4 * H_DIM;            // 384 x 96
    _Float16* s_x    = s_whh1 + G4 * H_DIM;            // 16 x 32 (K padded)
    _Float16* s_h1   = s_x + 16 * 32;                  // 16 x 96
    _Float16* s_h2   = s_h1 + 16 * H_DIM;              // 16 x 96
    float*    s_b0   = (float*)(s_h2 + 16 * H_DIM);    // 384 (bih0+bhh0)
    float*    s_b1   = s_b0 + G4;                      // 384
    float*    s_g    = s_b1 + G4;                      // 16 x 384 gate pre-acts

    const int tid = threadIdx.x;
    const int b0  = blockIdx.x * 16;

    // ---- one-time: stage weights fp32 -> fp16 LDS, fold biases ----
    for (int i = tid; i < G4 * 32; i += 256) {
        int n = i >> 5, k = i & 31;
        s_wih0[i] = (k < D_IN) ? (_Float16)Wih0[n * D_IN + k] : (_Float16)0.0f;
    }
    for (int i = tid; i < G4 * H_DIM; i += 256) {
        s_whh0[i] = (_Float16)Whh0[i];
        s_wih1[i] = (_Float16)Wih1[i];
        s_whh1[i] = (_Float16)Whh1[i];
    }
    if (tid < G4) {
        s_b0[tid] = bih0[tid] + bhh0[tid];
        s_b1[tid] = bih1[tid] + bhh1[tid];
    }
    for (int i = tid; i < 16 * 32; i += 256) s_x[i] = (_Float16)0.0f;
    for (int i = tid; i < 16 * H_DIM; i += 256) {
        s_h1[i] = (_Float16)0.0f;
        s_h2[i] = (_Float16)0.0f;
    }

    float c0r[6], c1r[6];
#pragma unroll
    for (int u = 0; u < 6; ++u) { c0r[u] = 0.0f; c1r[u] = 0.0f; }

    const int wave  = tid >> 5;
    const int lane  = tid & 31;
    const int kg    = lane >> 4;        // K half selector for A/B fragments
    const int nl    = lane & 15;        // N / M lane index
    const int mbase = kg * 8;           // C/D row base
    const int em    = (tid * 6) / H_DIM;   // elementwise: row
    const int ej    = (tid * 6) % H_DIM;   // elementwise: first col
    const int xm    = tid >> 4, xd = tid & 15;

    __syncthreads();

    // ---- hoist loop-invariant weight B-fragments into registers ----
    // 30 fragments x 8 VGPRs = 240 VGPRs/lane resident across the time loop.
    v16h wb_ih0[3];        // x -> gates, K=32 (padded)
    v16h wb_hh0[3][3];     // h1 -> gates, [tile][k-chunk]
    v16h wb_ih1[3][3];     // h1 -> gates (layer 1)
    v16h wb_hh1[3][3];     // h2 -> gates (layer 1)
    float bv0[3], bv1[3];  // per-lane gate bias (scalar, added at store time)
#pragma unroll
    for (int j = 0; j < 3; ++j) {
        const int row = (wave * 3 + j) * 16 + nl;       // weight row n = gate col
        wb_ih0[j] = ld_b(s_wih0 + row * 32, kg);
#pragma unroll
        for (int kk = 0; kk < 3; ++kk) {
            wb_hh0[j][kk] = ld_b(s_whh0 + row * H_DIM + 32 * kk, kg);
            wb_ih1[j][kk] = ld_b(s_wih1 + row * H_DIM + 32 * kk, kg);
            wb_hh1[j][kk] = ld_b(s_whh1 + row * H_DIM + 32 * kk, kg);
        }
        bv0[j] = s_b0[row];
        bv1[j] = s_b1[row];
    }

    const v8f vzero = {};

    for (int t = 0; t < T_LEN; ++t) {
        // ---- stage x_t tile (16x16 fp32 -> fp16, K padded to 32) ----
        {
            const float* xp = x + (((size_t)(b0 + xm)) * T_LEN + t) * D_IN + xd;
            s_x[xm * 32 + xd] = (_Float16)(*xp);
            if (t + 1 < T_LEN) __builtin_prefetch(xp + D_IN, 0, 0);  // global_prefetch_b8
        }
        __syncthreads();

        // ==== layer 0 matmul: gates = b0 + x@Wih0^T + h1@Whh0^T ====
        {
            v16h ax = ld_a(s_x + nl * 32, kg);
            v16h ah[3];
#pragma unroll
            for (int kk = 0; kk < 3; ++kk)
                ah[kk] = ld_a(s_h1 + nl * H_DIM + 32 * kk, kg);

            v8f acc[3];
            // first WMMA of each chain uses inline-zero C (no init movs)
#pragma unroll
            for (int j = 0; j < 3; ++j)
                acc[j] = wmma16(ax, wb_ih0[j], vzero);
#pragma unroll
            for (int kk = 0; kk < 3; ++kk)
#pragma unroll
                for (int j = 0; j < 3; ++j)
                    acc[j] = wmma16(ah[kk], wb_hh0[j][kk], acc[j]);

#pragma unroll
            for (int j = 0; j < 3; ++j) {
                const int n0 = (wave * 3 + j) * 16;
#pragma unroll
                for (int r = 0; r < 8; ++r)
                    s_g[(r + mbase) * G4 + n0 + nl] = acc[j][r] + bv0[j];
            }
        }
        __syncthreads();

        // ==== layer 0 elementwise: c,h update (6 cells/thread) ====
#pragma unroll
        for (int u = 0; u < 6; ++u) {
            const int j = ej + u;
            const float gi = s_g[em * G4 + j];
            const float gf = s_g[em * G4 + H_DIM + j];
            const float gg = s_g[em * G4 + 2 * H_DIM + j];
            const float go = s_g[em * G4 + 3 * H_DIM + j];
            const float c  = fast_sigmoid(gf) * c0r[u] + fast_sigmoid(gi) * fast_tanh(gg);
            c0r[u] = c;
            s_h1[em * H_DIM + j] = (_Float16)(fast_sigmoid(go) * fast_tanh(c));
        }
        __syncthreads();

        // ==== layer 1 matmul: gates = b1 + h1@Wih1^T + h2@Whh1^T ====
        {
            v16h a1[3], a2[3];
#pragma unroll
            for (int kk = 0; kk < 3; ++kk) {
                a1[kk] = ld_a(s_h1 + nl * H_DIM + 32 * kk, kg);
                a2[kk] = ld_a(s_h2 + nl * H_DIM + 32 * kk, kg);
            }

            v8f acc[3];
#pragma unroll
            for (int j = 0; j < 3; ++j)
                acc[j] = wmma16(a1[0], wb_ih1[j][0], vzero);
#pragma unroll
            for (int kk = 1; kk < 3; ++kk)
#pragma unroll
                for (int j = 0; j < 3; ++j)
                    acc[j] = wmma16(a1[kk], wb_ih1[j][kk], acc[j]);
#pragma unroll
            for (int kk = 0; kk < 3; ++kk)
#pragma unroll
                for (int j = 0; j < 3; ++j)
                    acc[j] = wmma16(a2[kk], wb_hh1[j][kk], acc[j]);

#pragma unroll
            for (int j = 0; j < 3; ++j) {
                const int n0 = (wave * 3 + j) * 16;
#pragma unroll
                for (int r = 0; r < 8; ++r)
                    s_g[(r + mbase) * G4 + n0 + nl] = acc[j][r] + bv1[j];
            }
        }
        __syncthreads();

        // ==== layer 1 elementwise ====
#pragma unroll
        for (int u = 0; u < 6; ++u) {
            const int j = ej + u;
            const float gi = s_g[em * G4 + j];
            const float gf = s_g[em * G4 + H_DIM + j];
            const float gg = s_g[em * G4 + 2 * H_DIM + j];
            const float go = s_g[em * G4 + 3 * H_DIM + j];
            const float c  = fast_sigmoid(gf) * c1r[u] + fast_sigmoid(gi) * fast_tanh(gg);
            c1r[u] = c;
            s_h2[em * H_DIM + j] = (_Float16)(fast_sigmoid(go) * fast_tanh(c));
        }
        __syncthreads();
    }

    // ==== FC head: out[b] = h2_last . Wfc + bfc ====
    if (tid < 16) {
        float s = bfc[0];
#pragma unroll 8
        for (int k = 0; k < H_DIM; ++k)
            s += (float)s_h2[tid * H_DIM + k] * Wfc[k];
        out[b0 + tid] = s;
    }
}

// ---------------------------------------------------------------------------
// LDS budget (bytes):
//   wih0 384*32*2 = 24576; whh0/wih1/whh1 = 3 * 384*96*2 = 221184
//   x 1024; h1/h2 3072 each; b0/b1 1536 each; gates 16*384*4 = 24576
//   total = 280576 B  (< 320KB per WGP)
// ---------------------------------------------------------------------------
static const int kLdsBytes = 280576;

extern "C" void kernel_launch(void* const* d_in, const int* in_sizes, int n_in,
                              void* d_out, int out_size, void* d_ws, size_t ws_size,
                              hipStream_t stream) {
    (void)in_sizes; (void)n_in; (void)out_size; (void)d_ws; (void)ws_size;
    const float* x    = (const float*)d_in[0];
    const float* Wih0 = (const float*)d_in[1];
    const float* Whh0 = (const float*)d_in[2];
    const float* bih0 = (const float*)d_in[3];
    const float* bhh0 = (const float*)d_in[4];
    const float* Wih1 = (const float*)d_in[5];
    const float* Whh1 = (const float*)d_in[6];
    const float* bih1 = (const float*)d_in[7];
    const float* bhh1 = (const float*)d_in[8];
    const float* Wfc  = (const float*)d_in[9];
    const float* bfc  = (const float*)d_in[10];
    float* out = (float*)d_out;

    // allow >64KB dynamic LDS (gfx1250 WGP has 320KB)
    hipFuncSetAttribute((const void*)BangaloreLSTM_kernel,
                        hipFuncAttributeMaxDynamicSharedMemorySize, kLdsBytes);

    hipLaunchKernelGGL(BangaloreLSTM_kernel,
                       dim3(B_TOT / 16), dim3(256), kLdsBytes, stream,
                       x, Wih0, Whh0, bih0, bhh0, Wih1, Whh1, bih1, bhh1,
                       Wfc, bfc, out);
}
